// GroupFNO1d_55929064129233
// MI455X (gfx1250) — matrix operand order repaired and checked by reading, not compiled, not run on hardware
//
#include <hip/hip_runtime.h>
#include <hip/hip_bf16.h>
#include <cstdint>
#include <cstddef>

// ---------------------------------------------------------------------------
// GroupFNO1d for MI455X (gfx1250, wave32, WMMA + Tensor Data Mover).
// spectral() with M=64 retained modes == two GEMMs against cos/sin bases +
// tiny per-(e,mode) complex weighting. All large matmuls -> bf16 WMMA GEMM.
// Tile staging uses TENSOR_LOAD_TO_LDS (TDM, 6-arg builtin on this toolchain).
// ---------------------------------------------------------------------------

typedef __attribute__((ext_vector_type(16))) __bf16 v16bf;
typedef __attribute__((ext_vector_type(8)))  float  v8f;
typedef __attribute__((ext_vector_type(4)))  unsigned int v4u;
typedef __attribute__((ext_vector_type(8)))  int    v8i;
typedef __attribute__((ext_vector_type(4)))  int    v4i;

static constexpr int kB  = 2048;
static constexpr int kE  = 16;
static constexpr int kS  = 512;
static constexpr int kW  = 512;
static constexpr int kNM = 64;        // retained modes
static constexpr int kC  = 64;
static constexpr int kO  = 8;
static constexpr int kR  = kB * kE;   // 32768 rows
static constexpr int kTW = 2 * kW;    // 1024 feature width
static constexpr int kFC = 128;       // 2*kNM spectral coef width

#if __has_builtin(__builtin_amdgcn_tensor_load_to_lds) && __has_builtin(__builtin_amdgcn_s_wait_tensorcnt)
#define USE_TDM 1
#else
#define USE_TDM 0
#endif

// ------------------------------ small kernels ------------------------------

__global__ void pack_input(const float* __restrict__ x, __bf16* __restrict__ H0) {
  size_t idx = (size_t)blockIdx.x * blockDim.x + threadIdx.x;
  if (idx >= (size_t)kR * kTW) return;
  int r = (int)(idx >> 10), c = (int)(idx & 1023);
  int b = r >> 4, e = r & 15;
  float v = (c < kS) ? x[(size_t)b * (kE * kS) + e * kS + c]
                     : (float)(c - kS) * (1.0f / (kS - 1));
  H0[idx] = (__bf16)v;
}

__global__ void cast_f32_bf16(const float* __restrict__ src, __bf16* __restrict__ dst, int n) {
  int i = blockIdx.x * blockDim.x + threadIdx.x;
  if (i < n) dst[i] = (__bf16)src[i];
}

// Forward DFT basis, K-major: Fb[2m][k]=cos, Fb[2m+1][k]=-sin
__global__ void build_fwd_basis(__bf16* __restrict__ Fb) {
  int idx = blockIdx.x * blockDim.x + threadIdx.x;      // kFC * kTW
  if (idx >= kFC * kTW) return;
  int row = idx >> 10, k = idx & 1023;
  int m = row >> 1;
  float ang = 6.28318530718f * (float)(m * k) / (float)kTW;
  float s, c; __sincosf(ang, &s, &c);
  Fb[idx] = (__bf16)((row & 1) ? -s : c);
}

// Inverse basis, K-major: Gt[t][2m]=s_m*cos, Gt[t][2m+1]=-s_m*sin
__global__ void build_inv_basis(__bf16* __restrict__ Gt) {
  int idx = blockIdx.x * blockDim.x + threadIdx.x;      // kTW * kFC
  if (idx >= kTW * kFC) return;
  int t = idx >> 7, q = idx & 127;
  int m = q >> 1;
  float sc = (m == 0) ? (1.0f / kTW) : (2.0f / kTW);
  float ang = 6.28318530718f * (float)(m * t) / (float)kTW;
  float s, c; __sincosf(ang, &s, &c);
  Gt[idx] = (__bf16)((q & 1) ? -sc * s : sc * c);
}

// wgt[e][m] = lam_m*com_m + (1-lam_m)*(codes[e,:] @ code[:,m])   (complex)
__global__ void mode_weights(const float* __restrict__ codes, const float* __restrict__ com,
                             const float* __restrict__ codew, const float* __restrict__ filt,
                             float2* __restrict__ wgt) {
  int idx = blockIdx.x * blockDim.x + threadIdx.x;      // kE * kNM
  if (idx >= kE * kNM) return;
  int e = idx / kNM, m = idx % kNM;
  float f = filt[m];
  float lam = fminf(fmaxf((f + 3.0f) * (1.0f / 6.0f), 0.0f), 1.0f);
  float er = 0.f, ei = 0.f;
  for (int c = 0; c < kC; ++c) {
    float cd = codes[e * kC + c];
    er += cd * codew[((size_t)c * kNM + m) * 2 + 0];
    ei += cd * codew[((size_t)c * kNM + m) * 2 + 1];
  }
  float wr = lam * com[m * 2 + 0] + (1.0f - lam) * er;
  float wi = lam * com[m * 2 + 1] + (1.0f - lam) * ei;
  wgt[idx] = make_float2(wr, wi);
}

// coefB[r][2m..2m+1] = wgt[r%16][m] * (coefF[r][2m] + i coefF[r][2m+1])
__global__ void apply_modes(const float* __restrict__ coefF, const float2* __restrict__ wgt,
                            __bf16* __restrict__ coefB) {
  size_t idx = (size_t)blockIdx.x * blockDim.x + threadIdx.x;   // kR * kNM
  if (idx >= (size_t)kR * kNM) return;
  int r = (int)(idx >> 6), m = (int)(idx & 63);
  int e = r & 15;
  float re = coefF[(size_t)r * kFC + 2 * m];
  float im = coefF[(size_t)r * kFC + 2 * m + 1];
  float2 w = wgt[e * kNM + m];
  coefB[(size_t)r * kFC + 2 * m]     = (__bf16)(re * w.x - im * w.y);
  coefB[(size_t)r * kFC + 2 * m + 1] = (__bf16)(re * w.y + im * w.x);
}

// grouped 2->4 matmul + bias + swish
__global__ void fc1_kernel(const __bf16* __restrict__ H3, const float* __restrict__ w,
                           const float* __restrict__ b, __bf16* __restrict__ H4) {
  size_t idx = (size_t)blockIdx.x * blockDim.x + threadIdx.x;   // kR * kW
  if (idx >= (size_t)kR * kW) return;
  int r = (int)(idx >> 9), g = (int)(idx & 511);
  float h0 = (float)H3[(size_t)r * kTW + 2 * g];
  float h1 = (float)H3[(size_t)r * kTW + 2 * g + 1];
#pragma unroll
  for (int o = 0; o < 4; ++o) {
    float v = h0 * w[(g * 4 + o) * 2 + 0] + h1 * w[(g * 4 + o) * 2 + 1] + b[g * 4 + o];
    v = v * (1.0f / (1.0f + __expf(-v)));
    H4[(size_t)r * 2048 + g * 4 + o] = (__bf16)v;
  }
}

// N=8 projection: weights staged in LDS; each thread streams one row once.
__global__ __launch_bounds__(256)
void fc2_kernel(const __bf16* __restrict__ H4, const __bf16* __restrict__ Wb,
                const float* __restrict__ bias, float* __restrict__ out) {
  __shared__ __bf16 sW[kO * 2048];     // 32 KB
  for (int i = threadIdx.x; i < kO * 2048 / 8; i += 256)
    *(uint4*)&sW[i * 8] = *(const uint4*)&Wb[i * 8];
  __syncthreads();

  const int r = blockIdx.x * 256 + threadIdx.x;      // kR/256 blocks, exact
  const __bf16* hp = H4 + (size_t)r * 2048;
  float acc[kO];
#pragma unroll
  for (int o = 0; o < kO; ++o) acc[o] = bias[o];
  for (int k = 0; k < 2048; k += 8) {
    union { uint4 u; __bf16 h[8]; } hv;
    hv.u = *(const uint4*)(hp + k);
#pragma unroll
    for (int o = 0; o < kO; ++o) {
      union { uint4 u; __bf16 w[8]; } wv;
      wv.u = *(const uint4*)&sW[o * 2048 + k];
      float a = acc[o];
#pragma unroll
      for (int u = 0; u < 8; ++u) a += (float)hv.h[u] * (float)wv.w[u];
      acc[o] = a;
    }
  }
  const int b = r >> 4, e = r & 15;
#pragma unroll
  for (int o = 0; o < kO; ++o) out[(size_t)b * (kE * kO) + e * kO + o] = acc[o];
}

// ------------------------- bf16 WMMA tiled GEMM ----------------------------
// C(M,N) = A(M,K) * Bt(N,K)^T [+bias] [+addM] [swish].  Tile 128x128, BK=32,
// 8 wave32 waves, each 32x64 (2x4 WMMA tiles). LDS rows strided by 48 bf16;
// TDM pad fields (16 DW interval, 8 DW pad) reproduce that stride in the DMA.

#define BM 128
#define BN 128
#define BKK 32
#define STR 48

#if USE_TDM
// Issue one 2D tile DMA: tile (tile_d1 rows x tile_d0 elems) of a bf16 matrix
// with row stride `stride_elems`, into LDS at lds_addr, padding 32B per 64B row.
__device__ __forceinline__ void tdm_load_tile_bf16(uint64_t gaddr, uint32_t lds_addr,
                                                   uint32_t tile_d0, uint32_t tile_d1,
                                                   uint64_t stride_elems,
                                                   uint32_t tdim0, uint32_t tdim1) {
  v4u g0;
  g0[0] = 1u;                                            // count=1, user D#
  g0[1] = lds_addr;                                      // LDS byte address
  g0[2] = (uint32_t)gaddr;                               // global_addr[31:0]
  g0[3] = (uint32_t)((gaddr >> 32) & 0x1FFFFFFu) | (2u << 30);  // [56:32] | type=2
  v8i g1;
  g1[0] = (int)((1u << 16) |                             // data_size = 2 bytes
                (1u << 20) |                             // pad_enable
                (3u << 22) |                             // pad_interval: 16 DWORDs
                (7u << 25));                             // pad_amount: 8 DWORDs
  g1[1] = (int)((tdim0 & 0xFFFFu) << 16);                // tensor_dim0[15:0]
  g1[2] = (int)(((tdim0 >> 16) & 0xFFFFu) | ((tdim1 & 0xFFFFu) << 16));
  g1[3] = (int)(((tdim1 >> 16) & 0xFFFFu) | ((tile_d0 & 0xFFFFu) << 16));
  g1[4] = (int)(tile_d1 & 0xFFFFu);                      // tile_dim1; tile_dim2=0
  g1[5] = (int)(uint32_t)(stride_elems & 0xFFFFFFFFu);   // dim0 stride lo
  g1[6] = (int)(uint32_t)((stride_elems >> 32) & 0xFFFFu);
  g1[7] = 0;
  v4i z4 = {0, 0, 0, 0};
  v8i z8 = {0, 0, 0, 0, 0, 0, 0, 0};
  // 6-arg form on this toolchain: (g0, g1, grp2, grp3, grp4, cpol)
  __builtin_amdgcn_tensor_load_to_lds(g0, g1, z4, z4, z8, 0);
}
#endif

template <bool HAS_BIAS, bool HAS_ADD, bool SWISH, bool OUT_BF>
__global__ __launch_bounds__(256)
void gemm_bf16_wmma(const __bf16* __restrict__ A, const __bf16* __restrict__ Bt,
                    const float* __restrict__ bias, const __bf16* __restrict__ addM,
                    __bf16* __restrict__ outB, float* __restrict__ outF,
                    int Mdim, int Ndim, int Kdim) {
  __shared__ __bf16 sA[2][BM * STR];
  __shared__ __bf16 sB[2][BN * STR];

  const int tid  = threadIdx.x;
  const int lane = tid & 31;
  const int wave = tid >> 5;
  const int wm   = wave & 3;
  const int wn   = wave >> 2;
  const int lh   = lane & 15;
  const int sel  = lane >> 4;
  const int m0   = blockIdx.x * BM;
  const int n0   = blockIdx.y * BN;

  v8f acc[2][4];
#pragma unroll
  for (int i = 0; i < 2; ++i)
#pragma unroll
    for (int j = 0; j < 4; ++j)
#pragma unroll
      for (int r = 0; r < 8; ++r) acc[i][j][r] = 0.0f;

#if USE_TDM
  const uint32_t ldsA = (uint32_t)(uintptr_t)&sA[0][0];
  const uint32_t ldsB = (uint32_t)(uintptr_t)&sB[0][0];
  auto issue_stage = [&](int buf, int kblk) {
    tdm_load_tile_bf16((uint64_t)(uintptr_t)(A + (size_t)m0 * Kdim + kblk),
                       ldsA + (uint32_t)buf * (BM * STR * 2),
                       BKK, BM, (uint64_t)Kdim, (uint32_t)Kdim, (uint32_t)Mdim);
    tdm_load_tile_bf16((uint64_t)(uintptr_t)(Bt + (size_t)n0 * Kdim + kblk),
                       ldsB + (uint32_t)buf * (BN * STR * 2),
                       BKK, BN, (uint64_t)Kdim, (uint32_t)Kdim, (uint32_t)Ndim);
  };
  if (wave == 0) {
    issue_stage(0, 0);
    __builtin_amdgcn_s_wait_tensorcnt(0);
  }
  __syncthreads();
#else
  const int c0 = tid, c1 = tid + 256;
  const int r0 = c0 >> 2, k0o = (c0 & 3) * 8;
  const int r1 = c1 >> 2, k1o = (c1 & 3) * 8;
  uint4 pa0, pa1, pb0, pb1;
  auto gload = [&](int kblk) {
    pa0 = *(const uint4*)(A  + (size_t)(m0 + r0) * Kdim + kblk + k0o);
    pa1 = *(const uint4*)(A  + (size_t)(m0 + r1) * Kdim + kblk + k1o);
    pb0 = *(const uint4*)(Bt + (size_t)(n0 + r0) * Kdim + kblk + k0o);
    pb1 = *(const uint4*)(Bt + (size_t)(n0 + r1) * Kdim + kblk + k1o);
  };
  auto sstore = [&](int buf) {
    *(uint4*)(&sA[buf][r0 * STR + k0o]) = pa0;
    *(uint4*)(&sA[buf][r1 * STR + k1o]) = pa1;
    *(uint4*)(&sB[buf][r0 * STR + k0o]) = pb0;
    *(uint4*)(&sB[buf][r1 * STR + k1o]) = pb1;
  };
  gload(0);
  sstore(0);
  __syncthreads();
#endif

  const int nk = Kdim / BKK;
  for (int s = 0; s < nk; ++s) {
    const int cur = s & 1;
    const bool more = (s + 1 < nk);
#if USE_TDM
    if (more && wave == 0) issue_stage(cur ^ 1, (s + 1) * BKK);   // DMA next tile
#else
    if (more) gload((s + 1) * BKK);
#endif

    // A fragment (16x32): lane<16 -> K{0..7,16..23}; lane>=16 -> K{8..15,24..31}
    union { uint4 u[2]; v16bf v; } af[2], bfr[4];
#pragma unroll
    for (int i = 0; i < 2; ++i) {
      const int row = wm * 32 + i * 16 + lh;
      af[i].u[0] = *(const uint4*)(&sA[cur][row * STR + sel * 8]);
      af[i].u[1] = *(const uint4*)(&sA[cur][row * STR + 16 + sel * 8]);
    }
    // B fragment (32x16): col N=lh; lane<16 -> K{0..7,16..23}; lane>=16 -> K{8..15,24..31}
#pragma unroll
    for (int j = 0; j < 4; ++j) {
      const int col = wn * 64 + j * 16 + lh;
      bfr[j].u[0] = *(const uint4*)(&sB[cur][col * STR + sel * 8]);
      bfr[j].u[1] = *(const uint4*)(&sB[cur][col * STR + 16 + sel * 8]);
    }
#pragma unroll
    for (int i = 0; i < 2; ++i)
#pragma unroll
      for (int j = 0; j < 4; ++j)
        acc[i][j] = __builtin_amdgcn_wmma_f32_16x16x32_bf16(
            false, af[i].v, false, bfr[j].v, (short)0, acc[i][j], false, false);

    if (more) {
#if USE_TDM
      if (wave == 0) __builtin_amdgcn_s_wait_tensorcnt(0);
#else
      sstore(cur ^ 1);
#endif
      __syncthreads();
    }
  }

  // epilogue (branch-free per template instantiation)
  float bcol[4];
#pragma unroll
  for (int j = 0; j < 4; ++j)
    bcol[j] = HAS_BIAS ? bias[n0 + wn * 64 + j * 16 + lh] : 0.0f;

#pragma unroll
  for (int i = 0; i < 2; ++i)
#pragma unroll
    for (int j = 0; j < 4; ++j)
#pragma unroll
      for (int r = 0; r < 8; ++r) {
        const int row = m0 + wm * 32 + i * 16 + sel * 8 + r;
        const int col = n0 + wn * 64 + j * 16 + lh;
        const size_t o = (size_t)row * Ndim + col;
        float v = acc[i][j][r] + bcol[j];
        if (HAS_ADD) v += (float)addM[o];
        if (SWISH)   v = v * (1.0f / (1.0f + __expf(-v)));
        if (OUT_BF)  outB[o] = (__bf16)v;
        else         outF[o] = v;
      }
}

// ------------------------------- launcher ----------------------------------

extern "C" void kernel_launch(void* const* d_in, const int* in_sizes, int n_in,
                              void* d_out, int out_size, void* d_ws, size_t ws_size,
                              hipStream_t stream) {
  (void)in_sizes; (void)n_in; (void)out_size; (void)ws_size;

  const float* x     = (const float*)d_in[0];
  const float* codes = (const float*)d_in[1];
  const float* p_w   = (const float*)d_in[2];
  const float* p_b   = (const float*)d_in[3];
  const float* com0  = (const float*)d_in[4];
  const float* code0 = (const float*)d_in[5];
  const float* filt0 = (const float*)d_in[6];
  const float* com1  = (const float*)d_in[7];
  const float* code1 = (const float*)d_in[8];
  const float* filt1 = (const float*)d_in[9];
  const float* w0_w  = (const float*)d_in[10];
  const float* w0_b  = (const float*)d_in[11];
  const float* w1_w  = (const float*)d_in[12];
  const float* w1_b  = (const float*)d_in[13];
  const float* fc1_w = (const float*)d_in[14];
  const float* fc1_b = (const float*)d_in[15];
  const float* fc2_w = (const float*)d_in[16];
  const float* fc2_b = (const float*)d_in[17];
  float* out = (float*)d_out;

  // ---- workspace layout (~222 MB; H4 aliases dead X1+H0 region) ----
  char* ws = (char*)d_ws;
  size_t off = 0;
  auto carve = [&](size_t bytes) { size_t p = off; off += (bytes + 255) & ~(size_t)255; return p; };
  float2* wgt0  = (float2*)(ws + carve((size_t)kE * kNM * sizeof(float2)));
  float2* wgt1  = (float2*)(ws + carve((size_t)kE * kNM * sizeof(float2)));
  __bf16* Fb    = (__bf16*)(ws + carve((size_t)kFC * kTW * 2));
  __bf16* Gt    = (__bf16*)(ws + carve((size_t)kTW * kFC * 2));
  __bf16* pwB   = (__bf16*)(ws + carve((size_t)kTW * kTW * 2));
  __bf16* w0B   = (__bf16*)(ws + carve((size_t)kTW * kTW * 2));
  __bf16* w1B   = (__bf16*)(ws + carve((size_t)kTW * kTW * 2));
  __bf16* fc2B  = (__bf16*)(ws + carve((size_t)kO * 2048 * 2));
  float*  coefF = (float*) (ws + carve((size_t)kR * kFC * 4));
  __bf16* coefB = (__bf16*)(ws + carve((size_t)kR * kFC * 2));
  __bf16* H1    = (__bf16*)(ws + carve((size_t)kR * kTW * 2));   // proj out (then H3)
  __bf16* X1    = (__bf16*)(ws + carve((size_t)kR * kTW * 2));   // spectral out
  __bf16* H0    = (__bf16*)(ws + carve((size_t)kR * kTW * 2));   // packed in (then H2)
  __bf16* H4    = X1;  // 128 MB, aliases X1 + H0 (both dead by fc1 time)

  const int T = 256;
  auto nb = [&](size_t n) { return (unsigned)((n + T - 1) / T); };

  mode_weights<<<nb(kE * kNM), T, 0, stream>>>(codes, com0, code0, filt0, wgt0);
  mode_weights<<<nb(kE * kNM), T, 0, stream>>>(codes, com1, code1, filt1, wgt1);
  build_fwd_basis<<<nb((size_t)kFC * kTW), T, 0, stream>>>(Fb);
  build_inv_basis<<<nb((size_t)kTW * kFC), T, 0, stream>>>(Gt);
  cast_f32_bf16<<<nb((size_t)kTW * kTW), T, 0, stream>>>(p_w,  pwB, kTW * kTW);
  cast_f32_bf16<<<nb((size_t)kTW * kTW), T, 0, stream>>>(w0_w, w0B, kTW * kTW);
  cast_f32_bf16<<<nb((size_t)kTW * kTW), T, 0, stream>>>(w1_w, w1B, kTW * kTW);
  cast_f32_bf16<<<nb((size_t)kO * 2048), T, 0, stream>>>(fc2_w, fc2B, kO * 2048);
  pack_input<<<nb((size_t)kR * kTW), T, 0, stream>>>(x, H0);

  dim3 blk(T);
  auto grd = [&](int Md, int Nd) { return dim3((unsigned)(Md / BM), (unsigned)(Nd / BN)); };

  // ---- layer 0 ----
  gemm_bf16_wmma<true, false, false, true><<<grd(kR, kTW), blk, 0, stream>>>(
      H0, pwB, p_b, nullptr, H1, nullptr, kR, kTW, kTW);            // h = proj
  gemm_bf16_wmma<false, false, false, false><<<grd(kR, kFC), blk, 0, stream>>>(
      H1, Fb, nullptr, nullptr, nullptr, coefF, kR, kFC, kTW);      // fwd modes
  apply_modes<<<nb((size_t)kR * kNM), T, 0, stream>>>(coefF, wgt0, coefB);
  gemm_bf16_wmma<false, false, false, true><<<grd(kR, kTW), blk, 0, stream>>>(
      coefB, Gt, nullptr, nullptr, X1, nullptr, kR, kTW, kFC);      // inverse
  gemm_bf16_wmma<true, true, true, true><<<grd(kR, kTW), blk, 0, stream>>>(
      H1, w0B, w0_b, X1, H0, nullptr, kR, kTW, kTW);                // H2 = swish(x1+x2)

  // ---- layer 1 ----
  gemm_bf16_wmma<false, false, false, false><<<grd(kR, kFC), blk, 0, stream>>>(
      H0, Fb, nullptr, nullptr, nullptr, coefF, kR, kFC, kTW);
  apply_modes<<<nb((size_t)kR * kNM), T, 0, stream>>>(coefF, wgt1, coefB);
  gemm_bf16_wmma<false, false, false, true><<<grd(kR, kTW), blk, 0, stream>>>(
      coefB, Gt, nullptr, nullptr, X1, nullptr, kR, kTW, kFC);
  gemm_bf16_wmma<true, true, false, true><<<grd(kR, kTW), blk, 0, stream>>>(
      H0, w1B, w1_b, X1, H1, nullptr, kR, kTW, kTW);                // H3 = x1+x2

  // ---- head ----
  fc1_kernel<<<nb((size_t)kR * kW), T, 0, stream>>>(H1, fc1_w, fc1_b, H4);
  fc2_kernel<<<(unsigned)(kR / 256), T, 0, stream>>>(H4, fc2B, fc2_b, out);
}
